// Model_29119878266993
// MI455X (gfx1250) — compile-verified, matrix operand order checked by reading, not run on hardware
//
#include <hip/hip_runtime.h>
#include <cstdint>
#include <cstddef>
#include <type_traits>

// ---------------------------------------------------------------------------
// Types matching CDNA5 WMMA builtin signatures (wave32, 16x16x32 bf16)
// ---------------------------------------------------------------------------
typedef __attribute__((ext_vector_type(16))) __bf16 v16bf;
typedef __attribute__((ext_vector_type(8)))  float  v8f;
typedef __attribute__((ext_vector_type(4)))  float  v4f;
typedef __attribute__((ext_vector_type(4)))  __bf16 bf16x4;

#define DD 128   // d_model

// A-source modes
enum { A_F32 = 0, A_BF16 = 1, A_CAT3 = 2, A_CAT2 = 3 };
// Epilogue modes
enum { E_BIAS_BF16 = 0, E_BIAS_F32 = 1, E_GLN_BF16 = 2,
       E_GLN_RES_F32 = 3, E_GLN_RES_SCAT = 4, E_GLN_F32 = 5 };

__device__ __forceinline__ float gelu_f(float x) {
  // jax.nn.gelu (approximate=True)
  float x3 = x * x * x;
  return 0.5f * x * (1.0f + tanhf(0.7978845608028654f * (x + 0.044715f * x3)));
}

// ---------------------------------------------------------------------------
// Fused GEMM: out[M x 128] = Epi( A[M x KT] @ W[KT x 128] + bias )
// Block: 256 threads = 8 wave32. Each WAVE owns a 32-row x 128-col strip
// (two 16-row groups): 16 accumulators v8f. Per K-step of 32:
//   2 A fragments + 8 B fragments -> 16 v_wmma_f32_16x16x32_bf16,
// each B fragment feeding both row groups; B column loop is software-
// pipelined (load c+1 before WMMA of c) for two B buffers in flight.
// KT compile-time: K-loop fully unrolls, every global_load_b128 is one
// per-lane base + immediate offset. No LDS, no barriers; LN via shfl_xor.
// ---------------------------------------------------------------------------
template <int AMODE, int EPI, int KT, bool GUARD>
__global__ __launch_bounds__(256) void gemm128(
    int M, int rowBase0,
    const float* __restrict__ Af32, const __bf16* __restrict__ Abf,
    const float* __restrict__ Esrc, const float* __restrict__ Hsrc,
    const int* __restrict__ senders, const int* __restrict__ recvs,
    const float* __restrict__ aggIn, const float* __restrict__ cnt,
    const __bf16* __restrict__ Wt,           // [128][KT] N-major bf16
    const float* __restrict__ bias,
    const float* __restrict__ lng, const float* __restrict__ lnb,
    const float* __restrict__ resIn,
    float* __restrict__ outF32, __bf16* __restrict__ outBf,
    float* __restrict__ aggOut)
{
  constexpr int K = KT;
  const int tid  = threadIdx.x;
  const int wave = tid >> 5, lane = tid & 31;
  const int lo = lane & 15, hi = lane >> 4;
  const int rowBase = rowBase0 + blockIdx.x * 256 + wave * 32;  // 32 rows per wave

  // per-row-group A-side rows (lane lo indexes the row within each group)
  int  arow[2];
  bool aOK[2];
  #pragma unroll
  for (int g = 0; g < 2; ++g) {
    arow[g] = rowBase + g * 16 + lo;
    aOK[g]  = GUARD ? (arow[g] < M) : true;
  }

  int sIdx[2] = {0, 0}, rIdx[2] = {0, 0};
  float invc[2] = {1.0f, 1.0f};
  #pragma unroll
  for (int g = 0; g < 2; ++g) {
    if constexpr (AMODE == A_CAT3) {
      if (aOK[g]) { sIdx[g] = senders[arow[g]]; rIdx[g] = recvs[arow[g]]; }
    }
    if constexpr (AMODE == A_CAT2) {
      if (aOK[g]) invc[g] = 1.0f / fmaxf(cnt[arow[g]], 1.0f);
    }
  }

  v8f acc[2][8];
  #pragma unroll
  for (int g = 0; g < 2; ++g)
    #pragma unroll
    for (int c = 0; c < 8; ++c) acc[g][c] = (v8f){0.f,0.f,0.f,0.f,0.f,0.f,0.f,0.f};

  // single per-lane base pointers; all loop offsets are compile-time immediates
  const __bf16* pb = Wt + (uint32_t)lo * K + (uint32_t)(hi * 8);
  const __bf16* paB[2] = {nullptr, nullptr};
  const float *paF[2] = {nullptr, nullptr}, *pe[2] = {nullptr, nullptr},
              *ps[2] = {nullptr, nullptr}, *pr[2] = {nullptr, nullptr},
              *ph[2] = {nullptr, nullptr}, *pg[2] = {nullptr, nullptr};
  #pragma unroll
  for (int g = 0; g < 2; ++g) {
    if constexpr (AMODE == A_BF16) paB[g] = Abf + (uint32_t)arow[g] * DD + (uint32_t)(hi * 8);
    if constexpr (AMODE == A_F32)  paF[g] = Af32 + (uint32_t)arow[g] * K + (uint32_t)(hi * 8);
    if constexpr (AMODE == A_CAT3) {
      pe[g] = Esrc + (uint32_t)arow[g] * DD + (uint32_t)(hi * 8);
      ps[g] = Hsrc + (uint32_t)sIdx[g] * DD + (uint32_t)(hi * 8);
      pr[g] = Hsrc + (uint32_t)rIdx[g] * DD + (uint32_t)(hi * 8);
    }
    if constexpr (AMODE == A_CAT2) {
      ph[g] = Hsrc  + (uint32_t)arow[g] * DD + (uint32_t)(hi * 8);
      pg[g] = aggIn + (uint32_t)arow[g] * DD + (uint32_t)(hi * 8);
    }
  }

  #pragma unroll
  for (int k0 = 0; k0 < K; k0 += 32) {
    // ---- two A fragments (one per 16-row group) ----
    v16bf af[2];
    #pragma unroll
    for (int g = 0; g < 2; ++g) {
      if (aOK[g]) {
        if constexpr (AMODE == A_BF16) {
          ((v4f*)&af[g])[0] = *(const v4f*)(paB[g] + k0);
          ((v4f*)&af[g])[1] = *(const v4f*)(paB[g] + k0 + 16);
        } else {
          v4f a[4];
          #pragma unroll
          for (int q = 0; q < 4; ++q) {
            const int c = k0 + (q >> 1) * 16 + (q & 1) * 4;   // compile-time; +hi*8 in base
            if constexpr (AMODE == A_F32) {
              a[q] = *(const v4f*)(paF[g] + c);
            } else if constexpr (AMODE == A_CAT3) {
              if (c < DD)          a[q] = *(const v4f*)(pe[g] + c);
              else if (c < 2 * DD) a[q] = *(const v4f*)(ps[g] + (c - DD));
              else                 a[q] = *(const v4f*)(pr[g] + (c - 2 * DD));
            } else { // A_CAT2
              if (c < DD) a[q] = *(const v4f*)(ph[g] + c);
              else { a[q] = *(const v4f*)(pg[g] + (c - DD)); a[q] *= invc[g]; }
            }
          }
          // packed f32->bf16 (lowers to v_cvt_pk_bf16_f32 pairs)
          #pragma unroll
          for (int q = 0; q < 4; ++q)
            ((bf16x4*)&af[g])[q] = __builtin_convertvector(a[q], bf16x4);
        }
      } else {
        v4f z = {0.f, 0.f, 0.f, 0.f};
        ((v4f*)&af[g])[0] = z; ((v4f*)&af[g])[1] = z;
      }
    }

    // ---- 8 column tiles, software-pipelined: load B(c+1) before WMMA(c) ----
    v16bf bcur, bnext;
    ((v4f*)&bcur)[0] = *(const v4f*)(pb + k0);
    ((v4f*)&bcur)[1] = *(const v4f*)(pb + k0 + 16);
    #pragma unroll
    for (int c = 0; c < 8; ++c) {
      if (c < 7) {
        ((v4f*)&bnext)[0] = *(const v4f*)(pb + ((c + 1) * 16 * K + k0));
        ((v4f*)&bnext)[1] = *(const v4f*)(pb + ((c + 1) * 16 * K + k0 + 16));
      }
      acc[0][c] = __builtin_amdgcn_wmma_f32_16x16x32_bf16(false, af[0], false, bcur,
                                                          (short)0, acc[0][c], false, false);
      acc[1][c] = __builtin_amdgcn_wmma_f32_16x16x32_bf16(false, af[1], false, bcur,
                                                          (short)0, acc[1][c], false, false);
      if (c < 7) bcur = bnext;
    }
  }

  // ---- epilogue: per group, C/D layout -> vgpr r = (row=g*16+hi*8+r, col=c*16+lo)
  float bv[8];
  #pragma unroll
  for (int c = 0; c < 8; ++c) bv[c] = bias[c * 16 + lo];

  if constexpr (EPI == E_BIAS_BF16 || EPI == E_BIAS_F32) {
    #pragma unroll
    for (int g = 0; g < 2; ++g) {
      #pragma unroll
      for (int r = 0; r < 8; ++r) {
        const int row = rowBase + g * 16 + hi * 8 + r;
        if (!GUARD || row < M) {
          #pragma unroll
          for (int c = 0; c < 8; ++c) {
            float x = acc[g][c][r] + bv[c];
            const uint32_t idx = (uint32_t)row * DD + (uint32_t)(c * 16 + lo);
            if constexpr (EPI == E_BIAS_BF16) outBf[idx] = (__bf16)x;
            else                              outF32[idx] = x;
          }
        }
      }
    }
  } else {
    float gv[8], b2[8];
    #pragma unroll
    for (int c = 0; c < 8; ++c) { gv[c] = lng[c * 16 + lo]; b2[c] = lnb[c * 16 + lo]; }

    #pragma unroll
    for (int g = 0; g < 2; ++g) {
      // gelu(x+bias), then per-row LayerNorm fully in registers
      #pragma unroll
      for (int c = 0; c < 8; ++c) {
        #pragma unroll
        for (int r = 0; r < 8; ++r) acc[g][c][r] = gelu_f(acc[g][c][r] + bv[c]);
      }
      float s1[8], s2[8];
      #pragma unroll
      for (int r = 0; r < 8; ++r) {
        float a0 = 0.f, a1 = 0.f;
        #pragma unroll
        for (int c = 0; c < 8; ++c) { float y = acc[g][c][r]; a0 += y; a1 += y * y; }
        s1[r] = a0; s2[r] = a1;
      }
      #pragma unroll
      for (int off = 1; off < 16; off <<= 1) {   // reduce within 16-lane halves
        #pragma unroll
        for (int r = 0; r < 8; ++r) {
          s1[r] += __shfl_xor(s1[r], off, 32);
          s2[r] += __shfl_xor(s2[r], off, 32);
        }
      }
      #pragma unroll
      for (int r = 0; r < 8; ++r) {
        const int row = rowBase + g * 16 + hi * 8 + r;
        if (GUARD && row >= M) continue;
        const float m   = s1[r] * (1.0f / DD);
        const float var = s2[r] * (1.0f / DD) - m * m;
        const float rs  = rsqrtf(var + 1e-5f);
        int rcv = 0;
        if constexpr (EPI == E_GLN_RES_SCAT) rcv = recvs[row];
        #pragma unroll
        for (int c = 0; c < 8; ++c) {
          float y = (acc[g][c][r] - m) * rs * gv[c] + b2[c];
          const uint32_t idx = (uint32_t)row * DD + (uint32_t)(c * 16 + lo);
          if constexpr (EPI == E_GLN_BF16) {
            outBf[idx] = (__bf16)y;
          } else if constexpr (EPI == E_GLN_F32) {
            outF32[idx] = y;
          } else {
            float z = resIn[idx] + y;
            outF32[idx] = z;
            if constexpr (EPI == E_GLN_RES_SCAT)
              atomicAdd(&aggOut[(uint32_t)rcv * DD + (uint32_t)(c * 16 + lo)], z);
          }
        }
      }
    }
  }
}

// ---------------------------------------------------------------------------
// Small helper kernels
// ---------------------------------------------------------------------------
// f32 [K][128] row-major  ->  bf16 [128][K] N-major (for direct B-fragment loads)
__global__ void k_cvt_t(const float* __restrict__ s, __bf16* __restrict__ d, int K) {
  int i = blockIdx.x * blockDim.x + threadIdx.x;
  if (i >= K * DD) return;
  int k = i / DD, n = i - k * DD;
  d[(size_t)n * K + k] = (__bf16)s[i];
}

__global__ void k_count(const int* __restrict__ recvs, float* __restrict__ cnt, int E) {
  int i = blockIdx.x * blockDim.x + threadIdx.x;
  if (i < E) atomicAdd(&cnt[recvs[i]], 1.0f);
}

__global__ void k_edge_emb(const float* __restrict__ edges, const float* __restrict__ w,
                           const float* __restrict__ b, float* __restrict__ e, int E) {
  int t = blockIdx.x * blockDim.x + threadIdx.x;   // E*32 threads, 4 cols each
  if (t >= E * 32) return;
  int ed = t >> 5;
  int c = (t & 31) << 2;
  float a0 = edges[(size_t)ed * 2 + 0];
  float a1 = edges[(size_t)ed * 2 + 1];
  v4f w0 = *(const v4f*)(w + c);
  v4f w1 = *(const v4f*)(w + DD + c);
  v4f bb = *(const v4f*)(b + c);
  v4f r = w0 * a0 + w1 * a1 + bb;
  *(v4f*)(e + (size_t)ed * DD + c) = r;
}

__global__ __launch_bounds__(256)
void k_proj(const float* __restrict__ h, const float* __restrict__ w,
            const float* __restrict__ b, float* __restrict__ out, int N) {
  int node = blockIdx.x * 8 + (threadIdx.x >> 5);   // one wave per node
  int lane = threadIdx.x & 31;
  if (node >= N || lane >= 24) return;
  const float* hr = h + (size_t)node * DD;
  float acc = b[lane];
  #pragma unroll 4
  for (int k = 0; k < DD; ++k) acc += hr[k] * w[k * 24 + lane];
  out[(size_t)node * 24 + lane] = acc;
}

// ---------------------------------------------------------------------------
// Host-side dispatch helper (full blocks unguarded + one guarded tail block)
// ---------------------------------------------------------------------------
template <int AM, int EP, int KT>
static void run_gemm(hipStream_t stream, int M, int full, int tail,
                     const float* Af32, const __bf16* Abf,
                     const float* Es, const float* Hs,
                     const int* sd, const int* rc,
                     const float* ag, const float* ct,
                     const __bf16* Wt, const float* bs,
                     const float* g, const float* bb,
                     const float* res, float* oF, __bf16* oB, float* aggO) {
  if (full > 0)
    gemm128<AM, EP, KT, false><<<full, 256, 0, stream>>>(
        M, 0, Af32, Abf, Es, Hs, sd, rc, ag, ct, Wt, bs, g, bb, res, oF, oB, aggO);
  if (tail > 0)
    gemm128<AM, EP, KT, true><<<1, 256, 0, stream>>>(
        M, full * 256, Af32, Abf, Es, Hs, sd, rc, ag, ct, Wt, bs, g, bb, res, oF, oB, aggO);
}

// ---------------------------------------------------------------------------
// Host launcher
// ---------------------------------------------------------------------------
extern "C" void kernel_launch(void* const* d_in, const int* in_sizes, int n_in,
                              void* d_out, int out_size, void* d_ws, size_t ws_size,
                              hipStream_t stream) {
  const float* nodes   = (const float*)d_in[0];
  const float* edges   = (const float*)d_in[1];
  const int*   senders = (const int*)d_in[2];
  const int*   recvs   = (const int*)d_in[3];
  const float* w_nemb  = (const float*)d_in[4];
  const float* b_nemb  = (const float*)d_in[5];
  const float* w_eemb  = (const float*)d_in[6];
  const float* b_eemb  = (const float*)d_in[7];
  const float* eproj_w = (const float*)d_in[8];
  const float* eproj_b = (const float*)d_in[9];
  const float* nproj_w = (const float*)d_in[10];
  const float* nproj_b = (const float*)d_in[11];
  const float* emlp_w  = (const float*)d_in[12];
  const float* emlp_b  = (const float*)d_in[13];
  const float* eln_g   = (const float*)d_in[14];
  const float* eln_b   = (const float*)d_in[15];
  const float* nmlp_w  = (const float*)d_in[16];
  const float* nmlp_b  = (const float*)d_in[17];
  const float* nln_g   = (const float*)d_in[18];
  const float* nln_b   = (const float*)d_in[19];
  const float* mout_w  = (const float*)d_in[20];
  const float* mout_b  = (const float*)d_in[21];
  const float* mout_g  = (const float*)d_in[22];
  const float* mout_be = (const float*)d_in[23];
  const float* proj_w  = (const float*)d_in[24];
  const float* proj_b  = (const float*)d_in[25];

  const int N = in_sizes[0] / (72 * 4);   // 20000
  const int E = in_sizes[2];              // 320000
  constexpr int KEMB = 72 * 4;            // 288

  // ---- workspace carve-out ----
  char* p = (char*)d_ws;
  auto alloc = [&](size_t bytes) -> char* {
    char* r = p; p += (bytes + 255) & ~(size_t)255; return r;
  };
  float*  h    = (float*) alloc((size_t)N * DD * 4);
  float*  e    = (float*) alloc((size_t)E * DD * 4);
  float*  agg  = (float*) alloc((size_t)N * DD * 4);
  float*  cnt  = (float*) alloc((size_t)N * 4);
  __bf16* t0   = (__bf16*)alloc((size_t)E * DD * 2);
  __bf16* t1   = (__bf16*)alloc((size_t)E * DD * 2);
  float*  hfin = (float*) alloc((size_t)N * DD * 4);
  __bf16* wb_nemb  = (__bf16*)alloc((size_t)KEMB * DD * 2);
  __bf16* wb_eproj = (__bf16*)alloc((size_t)2 * 3 * DD * DD * 2);
  __bf16* wb_nproj = (__bf16*)alloc((size_t)2 * 2 * DD * DD * 2);
  __bf16* wb_emlp  = (__bf16*)alloc((size_t)4 * DD * DD * 2);
  __bf16* wb_nmlp  = (__bf16*)alloc((size_t)4 * DD * DD * 2);
  __bf16* wb_mout  = (__bf16*)alloc((size_t)DD * DD * 2);

  // weight convert + transpose to N-major [128][K]
  auto cvtT = [&](const float* src, __bf16* dst, int K) {
    k_cvt_t<<<(K * DD + 255) / 256, 256, 0, stream>>>(src, dst, K);
  };
  cvtT(w_nemb, wb_nemb, KEMB);
  for (int l = 0; l < 2; ++l) {
    cvtT(eproj_w + (size_t)l * 3 * DD * DD, wb_eproj + (size_t)l * 3 * DD * DD, 3 * DD);
    cvtT(nproj_w + (size_t)l * 2 * DD * DD, wb_nproj + (size_t)l * 2 * DD * DD, 2 * DD);
  }
  for (int i = 0; i < 4; ++i) {
    cvtT(emlp_w + (size_t)i * DD * DD, wb_emlp + (size_t)i * DD * DD, DD);
    cvtT(nmlp_w + (size_t)i * DD * DD, wb_nmlp + (size_t)i * DD * DD, DD);
  }
  cvtT(mout_w, wb_mout, DD);

  hipMemsetAsync(cnt, 0, (size_t)N * 4, stream);
  k_count<<<(E + 255) / 256, 256, 0, stream>>>(recvs, cnt, E);

  k_edge_emb<<<(unsigned)((E * 32 + 255) / 256), 256, 0, stream>>>(edges, w_eemb, b_eemb, e, E);

  // 256 rows per block
  const int fullE = E / 256, tailE = E - fullE * 256;
  const int fullN = N / 256, tailN = N - fullN * 256;

  // h = nodes @ W_emb + b
  run_gemm<A_F32, E_BIAS_F32, KEMB>(stream, N, fullN, tailN,
      nodes, nullptr, nullptr, nullptr, nullptr, nullptr, nullptr, nullptr,
      wb_nemb, b_nemb, nullptr, nullptr, nullptr, h, nullptr, nullptr);

  for (int l = 0; l < 2; ++l) {
    hipMemsetAsync(agg, 0, (size_t)N * DD * 4, stream);
    // t0 = [e, h[s], h[r]] @ Wp + bp      (fused gather, bf16 out)
    run_gemm<A_CAT3, E_BIAS_BF16, 3 * DD>(stream, E, fullE, tailE,
        nullptr, nullptr, e, h, senders, recvs, nullptr, nullptr,
        wb_eproj + (size_t)l * 3 * DD * DD, eproj_b + l * DD,
        nullptr, nullptr, nullptr, nullptr, t0, nullptr);
    // t1 = LN(gelu(t0 @ W0 + b0))
    run_gemm<A_BF16, E_GLN_BF16, DD>(stream, E, fullE, tailE,
        nullptr, t0, nullptr, nullptr, nullptr, nullptr, nullptr, nullptr,
        wb_emlp + (size_t)(l * 2 + 0) * DD * DD, emlp_b + (l * 2 + 0) * DD,
        eln_g + (l * 2 + 0) * DD, eln_b + (l * 2 + 0) * DD,
        nullptr, nullptr, t1, nullptr);
    // e += LN(gelu(t1 @ W1 + b1));  agg[recv] += e   (fused scatter-sum)
    run_gemm<A_BF16, E_GLN_RES_SCAT, DD>(stream, E, fullE, tailE,
        nullptr, t1, nullptr, nullptr, nullptr, recvs, nullptr, nullptr,
        wb_emlp + (size_t)(l * 2 + 1) * DD * DD, emlp_b + (l * 2 + 1) * DD,
        eln_g + (l * 2 + 1) * DD, eln_b + (l * 2 + 1) * DD,
        e, e, nullptr, agg);
    // u0 = [h, agg/cnt] @ Wnp + bnp       (fused mean-normalize)
    run_gemm<A_CAT2, E_BIAS_BF16, 2 * DD>(stream, N, fullN, tailN,
        nullptr, nullptr, nullptr, h, nullptr, nullptr, agg, cnt,
        wb_nproj + (size_t)l * 2 * DD * DD, nproj_b + l * DD,
        nullptr, nullptr, nullptr, nullptr, t0, nullptr);
    // u1 = LN(gelu(u0 @ W0 + b0))
    run_gemm<A_BF16, E_GLN_BF16, DD>(stream, N, fullN, tailN,
        nullptr, t0, nullptr, nullptr, nullptr, nullptr, nullptr, nullptr,
        wb_nmlp + (size_t)(l * 2 + 0) * DD * DD, nmlp_b + (l * 2 + 0) * DD,
        nln_g + (l * 2 + 0) * DD, nln_b + (l * 2 + 0) * DD,
        nullptr, nullptr, t1, nullptr);
    // h += LN(gelu(u1 @ W1 + b1))
    run_gemm<A_BF16, E_GLN_RES_F32, DD>(stream, N, fullN, tailN,
        nullptr, t1, nullptr, nullptr, nullptr, nullptr, nullptr, nullptr,
        wb_nmlp + (size_t)(l * 2 + 1) * DD * DD, nmlp_b + (l * 2 + 1) * DD,
        nln_g + (l * 2 + 1) * DD, nln_b + (l * 2 + 1) * DD,
        h, h, nullptr, nullptr);
  }

  // hfin = LN(gelu(h @ W_out + b_out))
  run_gemm<A_F32, E_GLN_F32, DD>(stream, N, fullN, tailN,
      h, nullptr, nullptr, nullptr, nullptr, nullptr, nullptr, nullptr,
      wb_mout, mout_b, mout_g, mout_be,
      nullptr, hfin, nullptr, nullptr);

  // out = hfin @ proj_w + proj_b  -> [N, 24, 1]
  k_proj<<<(N + 7) / 8, 256, 0, stream>>>(hfin, proj_w, proj_b, (float*)d_out, N);
}
